// MoELayer_69303592288907
// MI455X (gfx1250) — matrix operand order repaired and checked by reading, not compiled, not run on hardware
//
#include <hip/hip_runtime.h>

// ---------------------------------------------------------------------------
// MoE layer for MI455X (gfx1250): bf16 WMMA sparse expert FFN.
// B=4, S=2048, D=1024, E=8, F=4096, k=2  -> T = 8192 tokens.
// Weights are fp32->bf16 converted AND transposed once per call so the
// WMMA GEMM hot loop does only aligned b128 LDS traffic (no scalar transpose).
// All 8 B fragments are loaded before the 8-WMMA burst so ds_load latency
// overlaps WMMA issue (forces distinct VGPR sets + partial dscnt waits).
// ---------------------------------------------------------------------------

typedef __attribute__((ext_vector_type(16))) __bf16 v16bf;
typedef __attribute__((ext_vector_type(8)))  float  v8f;

static constexpr int Bc = 4, Sc = 2048, Dc = 1024, Ec = 8, Fc = 4096;
static constexpr int Tc = Bc * Sc;          // 8192 tokens
static constexpr int TM = 128, TN = 128, TK = 32;
static constexpr int LSTR = 48;             // LDS row stride in halves (96B, 16B aligned)

union Frag16 {
    uint4 u[2];
    v16bf v;
    unsigned short s[16];
};

__device__ __forceinline__ unsigned short f2bf(float f) {
    unsigned int u = __float_as_uint(f);
    unsigned int r = (u + 0x7FFFu + ((u >> 16) & 1u)) >> 16;  // RNE
    return (unsigned short)r;
}

__device__ __forceinline__ float gelu_tanh(float x) {
    const float c0 = 0.7978845608028654f;   // sqrt(2/pi)
    float x3 = x * x * x;
    return 0.5f * x * (1.0f + tanhf(c0 * (x + 0.044715f * x3)));
}

// ---------------------------------------------------------------------------
// fp32 -> bf16 conversion (vectorized, grid-stride) -- for x only
// ---------------------------------------------------------------------------
__global__ __launch_bounds__(256) void cvt_bf16(const float* __restrict__ in,
                                                unsigned short* __restrict__ out,
                                                size_t n4) {
    size_t i = (size_t)blockIdx.x * blockDim.x + threadIdx.x;
    size_t stride = (size_t)gridDim.x * blockDim.x;
    for (; i < n4; i += stride) {
        float4 v = ((const float4*)in)[i];
        ushort4 r;
        r.x = f2bf(v.x); r.y = f2bf(v.y); r.z = f2bf(v.z); r.w = f2bf(v.w);
        ((ushort4*)out)[i] = r;
    }
}

// ---------------------------------------------------------------------------
// fp32 [E][K][N] -> bf16 [E][N][K] transpose-convert, 64x64 tiles via LDS.
// ---------------------------------------------------------------------------
__global__ __launch_bounds__(256) void cvt_bf16_transpose(
        const float* __restrict__ in, unsigned short* __restrict__ out,
        int K, int N) {
    __shared__ unsigned short lt[64][72];   // +8 halves pad: bank spread
    const size_t mat = (size_t)K * N;
    const float* src = in + (size_t)blockIdx.z * mat;
    unsigned short* dst = out + (size_t)blockIdx.z * mat;
    const int n0 = blockIdx.x * 64;
    const int k0 = blockIdx.y * 64;
    const int t = threadIdx.x;
    {
        const int r   = t >> 2;             // k-local 0..63
        const int seg = t & 3;              // 16-float segment
        const float4* p = (const float4*)(src + (size_t)(k0 + r) * N + n0 + seg * 16);
#pragma unroll
        for (int q = 0; q < 4; ++q) {
            float4 v = p[q];
            int c = seg * 16 + q * 4;
            lt[r][c + 0] = f2bf(v.x); lt[r][c + 1] = f2bf(v.y);
            lt[r][c + 2] = f2bf(v.z); lt[r][c + 3] = f2bf(v.w);
        }
    }
    __syncthreads();
    {
        const int n   = t >> 2;             // n-local 0..63
        const int seg = t & 3;              // k quarter
        __align__(16) unsigned short tmp[16];
#pragma unroll
        for (int q = 0; q < 16; ++q) tmp[q] = lt[seg * 16 + q][n];
        uint4* o = (uint4*)(dst + (size_t)(n0 + n) * K + k0 + seg * 16);
        o[0] = *(const uint4*)&tmp[0];
        o[1] = *(const uint4*)&tmp[8];
    }
}

// ---------------------------------------------------------------------------
// Gate: logits = x @ Wg [T,8], top-2 softmax, append to per-expert lists.
// ---------------------------------------------------------------------------
__global__ __launch_bounds__(256) void gate_kernel(const float* __restrict__ x,
                                                   const float* __restrict__ Wg,
                                                   int* __restrict__ cnt,
                                                   int* __restrict__ list,
                                                   float* __restrict__ wgt) {
    const int lane = threadIdx.x & 31;
    const int wv   = threadIdx.x >> 5;
    const int t    = blockIdx.x * 8 + wv;

    float acc[Ec];
#pragma unroll
    for (int e = 0; e < Ec; ++e) acc[e] = 0.0f;

    const float* xr = x + (size_t)t * Dc;
    for (int d = lane; d < Dc; d += 32) {
        float xv = xr[d];
        const float4* wg = (const float4*)(Wg + (size_t)d * Ec);
        float4 w0 = wg[0], w1 = wg[1];
        acc[0] += xv * w0.x; acc[1] += xv * w0.y;
        acc[2] += xv * w0.z; acc[3] += xv * w0.w;
        acc[4] += xv * w1.x; acc[5] += xv * w1.y;
        acc[6] += xv * w1.z; acc[7] += xv * w1.w;
    }
#pragma unroll
    for (int off = 16; off > 0; off >>= 1) {
#pragma unroll
        for (int e = 0; e < Ec; ++e)
            acc[e] += __shfl_xor(acc[e], off, 32);
    }
    if (lane == 0) {
        int   i0 = 0;  float v0 = acc[0];
#pragma unroll
        for (int e = 1; e < Ec; ++e)
            if (acc[e] > v0) { v0 = acc[e]; i0 = e; }     // ties -> lowest index
        int   i1 = -1; float v1 = -3.4e38f;
#pragma unroll
        for (int e = 0; e < Ec; ++e)
            if (e != i0 && acc[e] > v1) { v1 = acc[e]; i1 = e; }
        float e1 = __expf(v1 - v0);
        float s  = 1.0f + e1;
        float w0 = 1.0f / s, w1 = e1 / s;
        int p0 = atomicAdd(&cnt[i0], 1);
        list[(size_t)i0 * Tc + p0] = t;  wgt[(size_t)i0 * Tc + p0] = w0;
        int p1 = atomicAdd(&cnt[i1], 1);
        list[(size_t)i1 * Tc + p1] = t;  wgt[(size_t)i1 * Tc + p1] = w1;
    }
}

// ---------------------------------------------------------------------------
// Tiled bf16 WMMA GEMM over one expert's routed tokens.
//   A [rows, K] bf16 (row-major), B' [N, K] bf16 (pre-transposed)
//   FFN1=true : C = gelu(gather(x) @ w1[e])  -> h (bf16)
//   FFN1=false: out[tok] += w * (h @ w2[e])  (fp32 RMW, single writer/launch)
// ---------------------------------------------------------------------------
template <bool FFN1>
__global__ __launch_bounds__(256) void moe_ffn_gemm(
        const unsigned short* __restrict__ Abase,   // [rows, K] bf16
        const unsigned short* __restrict__ Bbase,   // [N, K]    bf16 (transposed)
        const int*   __restrict__ list,
        const float* __restrict__ wgt,
        const int*   __restrict__ cntPtr,
        unsigned short* __restrict__ hOut,
        float*          __restrict__ outBuf,
        int K, int N) {
    const int cnt = cntPtr[0];
    const int rowBase = blockIdx.y * TM;
    if (rowBase >= cnt) return;                     // uniform early exit
    const int n0 = blockIdx.x * TN;

    __shared__ unsigned short lA[TM * LSTR];
    __shared__ unsigned short lB[TN * LSTR];
    __shared__ int   lTok[TM];
    __shared__ float lW[TM];

    const int tid = threadIdx.x;
    if (tid < TM) {
        int pos = rowBase + tid;
        bool v = pos < cnt;
        lTok[tid] = v ? list[pos] : 0;
        if constexpr (!FFN1) lW[tid] = v ? wgt[pos] : 0.0f;
    }
    __syncthreads();

    // ---- loader roles (identical pattern for A rows and B' rows) ----
    const int lr = tid & (TM - 1);                  // A row / B' row 0..127
    const int lh = tid >> 7;                        // K-half 0/1 (16 halves each)

    size_t arow;
    if constexpr (FFN1) arow = (size_t)lTok[lr] * (size_t)K;   // gather token row
    else                arow = (size_t)(rowBase + lr) * (size_t)K;
    const size_t brow = (size_t)(n0 + lr) * (size_t)K;

    // ---- wave/fragment roles ----
    const int lane = tid & 31;
    const int wv   = tid >> 5;       // wave 0..7 -> rows wv*16..wv*16+15
    const int kh   = lane >> 4;      // lane half
    const int m16  = lane & 15;
    const int am   = wv * 16 + m16;  // A fragment row

    v8f zero = {};
    v8f acc[8];
#pragma unroll
    for (int j = 0; j < 8; ++j) acc[j] = zero;

    // ---- prologue: stage k-tile 0 into registers ----
    uint4 a0, a1, b0, b1;
    {
        const uint4* ap = (const uint4*)(Abase + arow + lh * 16);
        a0 = ap[0]; a1 = ap[1];
        const uint4* bp = (const uint4*)(Bbase + brow + lh * 16);
        b0 = bp[0]; b1 = bp[1];
    }

    const int nK = K / TK;
    for (int kt = 0; kt < nK; ++kt) {
        __syncthreads();   // previous compute done reading LDS
        *(uint4*)&lA[lr * LSTR + lh * 16]     = a0;
        *(uint4*)&lA[lr * LSTR + lh * 16 + 8] = a1;
        *(uint4*)&lB[lr * LSTR + lh * 16]     = b0;
        *(uint4*)&lB[lr * LSTR + lh * 16 + 8] = b1;
        __syncthreads();

        // stage next tile while computing on this one
        if (kt + 1 < nK) {
            const uint4* ap = (const uint4*)(Abase + arow + (size_t)(kt + 1) * TK + lh * 16);
            a0 = ap[0]; a1 = ap[1];
            const uint4* bp = (const uint4*)(Bbase + brow + (size_t)(kt + 1) * TK + lh * 16);
            b0 = bp[0]; b1 = bp[1];
        }
        // prefetch k-tile+2 into caches (global_prefetch_b8)
        if (kt + 2 < nK) {
            __builtin_prefetch(Abase + arow + (size_t)(kt + 2) * TK + lh * 16, 0, 1);
            __builtin_prefetch(Bbase + brow + (size_t)(kt + 2) * TK + lh * 16, 0, 1);
        }

        // A fragment: 16-bit A 16x32 layout (lanes 0-15: K 0..7,16..23; 16-31: +8)
        Frag16 fa;
        fa.u[0] = *(const uint4*)&lA[am * LSTR + kh * 8];
        fa.u[1] = *(const uint4*)&lA[am * LSTR + 16 + kh * 8];

        // Load ALL 8 B fragments first (simultaneously live -> distinct VGPRs,
        // partial dscnt waits), then issue the 8-WMMA burst back-to-back.
        Frag16 fb[8];
#pragma unroll
        for (int j = 0; j < 8; ++j) {
            const int n = j * 16 + m16;
            fb[j].u[0] = *(const uint4*)&lB[n * LSTR + kh * 8];
            fb[j].u[1] = *(const uint4*)&lB[n * LSTR + 16 + kh * 8];
        }
#pragma unroll
        for (int j = 0; j < 8; ++j) {
            acc[j] = __builtin_amdgcn_wmma_f32_16x16x32_bf16(
                false, fa.v, false, fb[j].v, (short)0, acc[j], false, false);
        }
    }

    // ---- epilogue: D layout, element r -> row wv*16 + kh*8 + r, col j*16 + m16
#pragma unroll
    for (int r = 0; r < 8; ++r) {
        const int rowLocal = wv * 16 + kh * 8 + r;
        const int pos = rowBase + rowLocal;
        if (pos < cnt) {
            if constexpr (FFN1) {
                unsigned short* dst = hOut + (size_t)pos * N;
#pragma unroll
                for (int j = 0; j < 8; ++j) {
                    const int col = n0 + j * 16 + m16;
                    dst[col] = f2bf(gelu_tanh(acc[j][r]));
                }
            } else {
                const float w = lW[rowLocal];
                float* dst = outBuf + (size_t)lTok[rowLocal] * N;
#pragma unroll
                for (int j = 0; j < 8; ++j) {
                    const int col = n0 + j * 16 + m16;
                    dst[col] += w * acc[j][r];      // single writer per launch
                }
            }
        }
    }
}

// ---------------------------------------------------------------------------
// Host-side launcher
// ---------------------------------------------------------------------------
extern "C" void kernel_launch(void* const* d_in, const int* in_sizes, int n_in,
                              void* d_out, int out_size, void* d_ws, size_t ws_size,
                              hipStream_t stream) {
    const float* x  = (const float*)d_in[0];   // [B,S,D]
    const float* Wg = (const float*)d_in[1];   // [D,E]
    const float* w1 = (const float*)d_in[2];   // [E,D,F]
    const float* w2 = (const float*)d_in[3];   // [E,F,D]
    (void)in_sizes; (void)n_in; (void)ws_size;

    // workspace carve-up (256B aligned)
    char* ws = (char*)d_ws;
    size_t off = 0;
    auto carve = [&](size_t bytes) -> void* {
        void* p = ws + off;
        off = (off + bytes + 255) & ~(size_t)255;
        return p;
    };
    int*            cnt  = (int*)carve(Ec * sizeof(int));
    int*            list = (int*)carve((size_t)Ec * Tc * sizeof(int));
    float*          wgt  = (float*)carve((size_t)Ec * Tc * sizeof(float));
    unsigned short* xbf  = (unsigned short*)carve((size_t)Tc * Dc * 2);
    unsigned short* w1t  = (unsigned short*)carve((size_t)Ec * Dc * Fc * 2); // [E][N=F][K=D]
    unsigned short* w2t  = (unsigned short*)carve((size_t)Ec * Fc * Dc * 2); // [E][N=D][K=F]
    unsigned short* h    = (unsigned short*)carve((size_t)Tc * Fc * 2);

    hipMemsetAsync(cnt, 0, Ec * sizeof(int), stream);
    hipMemsetAsync(d_out, 0, (size_t)out_size * sizeof(float), stream);

    // x: fp32 -> bf16 (layout kept)
    cvt_bf16<<<2048, 256, 0, stream>>>(x, xbf, (size_t)Tc * Dc / 4);
    // w1 [E][D][F] -> bf16 [E][F][D];  w2 [E][F][D] -> bf16 [E][D][F]
    cvt_bf16_transpose<<<dim3(Fc / 64, Dc / 64, Ec), 256, 0, stream>>>(w1, w1t, Dc, Fc);
    cvt_bf16_transpose<<<dim3(Dc / 64, Fc / 64, Ec), 256, 0, stream>>>(w2, w2t, Fc, Dc);

    // gate + routing lists
    gate_kernel<<<Tc / 8, 256, 0, stream>>>(x, Wg, cnt, list, wgt);

    // per-expert sparse FFN (sequential -> h buffer reuse, no output atomics)
    for (int e = 0; e < Ec; ++e) {
        const unsigned short* w1e = w1t + (size_t)e * Dc * Fc;
        const unsigned short* w2e = w2t + (size_t)e * Fc * Dc;
        const int*   le = list + (size_t)e * Tc;
        const float* we = wgt  + (size_t)e * Tc;

        dim3 g1(Fc / TN, Tc / TM);   // 32 x 64
        moe_ffn_gemm<true><<<g1, 256, 0, stream>>>(
            xbf, w1e, le, we, cnt + e, h, nullptr, Dc, Fc);

        dim3 g2(Dc / TN, Tc / TM);   // 8 x 64
        moe_ffn_gemm<false><<<g2, 256, 0, stream>>>(
            h, w2e, le, we, cnt + e, nullptr, (float*)d_out, Fc, Dc);
    }
}
